// PiNet_12532714569877
// MI455X (gfx1250) — matrix operand order repaired and checked by reading, not compiled, not run on hardware
//
#include <hip/hip_runtime.h>
#include <hip/hip_bf16.h>

typedef _Float16 v16h __attribute__((ext_vector_type(16)));
typedef _Float16 v8h  __attribute__((ext_vector_type(8)));
typedef float    v8f  __attribute__((ext_vector_type(8)));

#define CUTOFF_F 5.0f

#if __has_builtin(__builtin_amdgcn_tanhf)
  #define TANH(x) __builtin_amdgcn_tanhf(x)
#elif __has_builtin(__builtin_amdgcn_tanh_f32)
  #define TANH(x) __builtin_amdgcn_tanh_f32(x)
#else
  #define TANH(x) tanhf(x)
#endif

#if __has_builtin(__builtin_amdgcn_exp2f)
  #define EXP2(x) __builtin_amdgcn_exp2f(x)
#else
  #define EXP2(x) exp2f(x)
#endif

// COS2PI(x) = cos(2*pi*x)   (v_cos_f32 operates in revolutions)
#if __has_builtin(__builtin_amdgcn_cosf)
  #define COS2PI(x) __builtin_amdgcn_cosf(x)
#else
  #define COS2PI(x) cosf((x) * 6.28318530717958647692f)
#endif

// ---------------------------------------------------------------------------
// Fragment-order weight staging.
//
// A-matrix 16x32 f16 layout (per ISA 7.12.2): lane L (hi=L>>4, lo=L&15) holds
// row m=lo; element j (0..15) is k = kh*32 + hi*8 + (j&7) + ((j>>3)<<4).
// B-matrix 32x16 f16 layout: lane L holds col n=lo; element j is
// k = kh*32 + hi*8 + (j&7) + ((j>>3)<<4)   (same k order as A).
//
// Weights W[k][n] (64x64 row-major) are pre-shuffled into LDS so that each
// lane's 16-element B fragment is contiguous; lane stride padded to 24
// elements (48B) -> 16B-aligned ds_load_b128, 2-way-max bank pattern.
// ---------------------------------------------------------------------------
#define WFRAG_LANE_STRIDE 24
#define WFRAG_MAT_ELEMS   (8 * 32 * WFRAG_LANE_STRIDE)   // (4 nt * 2 kh) * 32 lanes * 24

__device__ __forceinline__ int wfrag_slot(int k, int n) {
    int kh   = k >> 5;
    int j    = (k & 7) + (((k >> 4) & 1) << 3);
    int hb   = (k >> 3) & 1;
    int lane = hb * 16 + (n & 15);
    int nt   = n >> 4;
    return ((nt * 2 + kh) * 32 + lane) * WFRAG_LANE_STRIDE + j;
}

__device__ __forceinline__ v16h load_bfrag(const _Float16* wmat, int nt, int kh, int lane) {
    const _Float16* wp = wmat + ((nt * 2 + kh) * 32 + lane) * WFRAG_LANE_STRIDE;
    v8h w0 = *(const v8h*)(wp);
    v8h w1 = *(const v8h*)(wp + 8);
    v16h b;
#pragma unroll
    for (int j = 0; j < 8; ++j) { b[j] = w0[j]; b[8 + j] = w1[j]; }
    return b;
}

// Per-wave [16][64] f16 staging tile; row stride 80 elems (160B, 16B aligned).
#define S_ROW 80
#define S_ELEMS (16 * S_ROW)

__device__ __forceinline__ v16h load_afrag_from_S(const _Float16* Sm, int lo, int hi, int kh) {
    const _Float16* p = Sm + lo * S_ROW + kh * 32 + hi * 8;
    v8h a0 = *(const v8h*)(p);
    v8h a1 = *(const v8h*)(p + 16);
    v16h a;
#pragma unroll
    for (int j = 0; j < 8; ++j) { a[j] = a0[j]; a[8 + j] = a1[j]; }
    return a;
}

#define WMMA(a, b, c) __builtin_amdgcn_wmma_f32_16x16x32_f16(false, (a), false, (b), (short)0, (c), false, false)

// ---------------------------------------------------------------------------
__global__ void pinet_zero(float* __restrict__ p, int n) {
    int i = blockIdx.x * 256 + threadIdx.x;
    if (i < n) p[i] = 0.0f;
}

// ---------------------------------------------------------------------------
// Node kernel: pp1 = tanh(tanh(emb[Z]@Wpp1+b1)@Wpp2+b2), one wave per 16 nodes.
// ---------------------------------------------------------------------------
__global__ __launch_bounds__(256) void pinet_node(
    const int* __restrict__ Z, const float* __restrict__ emb,
    const float* __restrict__ Wpp1, const float* __restrict__ bpp1,
    const float* __restrict__ Wpp2, const float* __restrict__ bpp2,
    float* __restrict__ pp1, int Nn, int Ntiles)
{
    __shared__ __align__(32) _Float16 wfrag[2][WFRAG_MAT_ELEMS];
    __shared__ __align__(32) _Float16 S[8][S_ELEMS];

    int tid = threadIdx.x;
    for (int e = tid; e < 64 * 64; e += 256) {
        int k = e >> 6, n = e & 63;
        int slot = wfrag_slot(k, n);
        wfrag[0][slot] = (_Float16)Wpp1[e];
        wfrag[1][slot] = (_Float16)Wpp2[e];
    }
    __syncthreads();

    int lane = tid & 31, wave = tid >> 5;
    int hi = lane >> 4, lo = lane & 15;
    int tile = blockIdx.x * 8 + wave;
    if (tile >= Ntiles) return;
    int base = tile * 16;
    _Float16* Sm = &S[wave][0];

    float b1v[4], b2v[4];
#pragma unroll
    for (int nt = 0; nt < 4; ++nt) { b1v[nt] = bpp1[nt * 16 + lo]; b2v[nt] = bpp2[nt * 16 + lo]; }

    // A fragments: gather embedding row for node m = lo of this tile.
    int node = base + lo; if (node >= Nn) node = Nn - 1;
    const float* erow = emb + (size_t)Z[node] * 64;
    v16h afrag[2];
#pragma unroll
    for (int kh = 0; kh < 2; ++kh) {
        int k0 = kh * 32 + hi * 8;
#pragma unroll
        for (int j = 0; j < 8; ++j) {
            afrag[kh][j]     = (_Float16)erow[k0 + j];
            afrag[kh][8 + j] = (_Float16)erow[k0 + 16 + j];
        }
    }

    v8f acc[4];
#pragma unroll
    for (int nt = 0; nt < 4; ++nt) {
        v8f c = {};
#pragma unroll
        for (int kh = 0; kh < 2; ++kh)
            c = WMMA(afrag[kh], load_bfrag(&wfrag[0][0], nt, kh, lane), c);
        acc[nt] = c;
    }

    // tanh + bias, transpose-stage through LDS (C/D layout -> A layout).
#pragma unroll
    for (int nt = 0; nt < 4; ++nt)
#pragma unroll
        for (int v = 0; v < 8; ++v)
            Sm[(v + 8 * hi) * S_ROW + nt * 16 + lo] = (_Float16)TANH(acc[nt][v] + b1v[nt]);
    __builtin_amdgcn_wave_barrier();   // same-wave DS ops are HW-ordered; stop compiler reordering

#pragma unroll
    for (int kh = 0; kh < 2; ++kh) afrag[kh] = load_afrag_from_S(Sm, lo, hi, kh);

#pragma unroll
    for (int nt = 0; nt < 4; ++nt) {
        v8f c = {};
#pragma unroll
        for (int kh = 0; kh < 2; ++kh)
            c = WMMA(afrag[kh], load_bfrag(&wfrag[1][0], nt, kh, lane), c);
        acc[nt] = c;
    }

    // Coalesced fp32 store of pp1 (lanes 0-15 -> 64B contiguous per (v,nt)).
#pragma unroll
    for (int nt = 0; nt < 4; ++nt)
#pragma unroll
        for (int v = 0; v < 8; ++v) {
            int m = base + v + 8 * hi;
            if (m < Nn) pp1[(size_t)m * 64 + nt * 16 + lo] = TANH(acc[nt][v] + b2v[nt]);
        }
}

// ---------------------------------------------------------------------------
// Edge kernel: fused basis -> phi -> inter -> PI -> II -> scatter_add.
// One wave per 16 edges. 24 WMMAs per tile.
// ---------------------------------------------------------------------------
__global__ __launch_bounds__(256) void pinet_edge(
    const float* __restrict__ R, const int* __restrict__ idx_i,
    const int* __restrict__ idx_j, const float* __restrict__ offs,
    const float* __restrict__ Wb,  const float* __restrict__ bb,
    const float* __restrict__ Wpi, const float* __restrict__ bpi,
    const float* __restrict__ Wii, const float* __restrict__ bii,
    const float* __restrict__ pp1, float* __restrict__ out,
    int Ecnt, int Etiles)
{
    __shared__ __align__(32) _Float16 wfrag[3][WFRAG_MAT_ELEMS];
    __shared__ __align__(32) _Float16 S[8][S_ELEMS];

    int tid = threadIdx.x;
    for (int e = tid; e < 64 * 64; e += 256) {
        int k = e >> 6, n = e & 63;
        int slot = wfrag_slot(k, n);
        wfrag[0][slot] = (_Float16)Wb[e];
        wfrag[1][slot] = (_Float16)Wpi[e];
        wfrag[2][slot] = (_Float16)Wii[e];
    }
    __syncthreads();

    int lane = tid & 31, wave = tid >> 5;
    int hi = lane >> 4, lo = lane & 15;
    int tile = blockIdx.x * 8 + wave;
    if (tile >= Etiles) return;
    int base = tile * 16;
    _Float16* Sm = &S[wave][0];

    float bbv[4], bpiv[4], biiv[4];
#pragma unroll
    for (int nt = 0; nt < 4; ++nt) {
        bbv[nt] = bb[nt * 16 + lo]; bpiv[nt] = bpi[nt * 16 + lo]; biiv[nt] = bii[nt * 16 + lo];
    }

    // ---- geometry for edge m = lo of this tile (duplicated across halves) ----
    int e0 = base + lo; if (e0 >= Ecnt) e0 = Ecnt - 1;
    int gi = idx_i[e0], gj = idx_j[e0];
    float dx = R[gj * 3 + 0] - R[gi * 3 + 0] + offs[e0 * 3 + 0];
    float dy = R[gj * 3 + 1] - R[gi * 3 + 1] + offs[e0 * 3 + 1];
    float dz = R[gj * 3 + 2] - R[gi * 3 + 2] + offs[e0 * 3 + 2];
    float d  = sqrtf(dx * dx + dy * dy + dz * dz);
    float fc = 0.5f * (COS2PI(d * (0.5f / CUTOFF_F)) + 1.0f) * (d < CUTOFF_F ? 1.0f : 0.0f);

    // ---- basis * fc directly into A-fragment registers ----
    const float step  = CUTOFF_F / 63.0f;      // linspace(0, cutoff, 64)
    const float inv_w = 64.0f / CUTOFF_F;      // 1/width
    v16h afrag[2];
#pragma unroll
    for (int kh = 0; kh < 2; ++kh) {
#pragma unroll
        for (int j = 0; j < 16; ++j) {
            int k = kh * 32 + hi * 8 + (j & 7) + ((j >> 3) << 4);
            float cc = (d - (float)k * step) * inv_w;
            afrag[kh][j] = (_Float16)(EXP2(-0.72134752044448170368f * cc * cc) * fc);
        }
    }

    // ---- GEMM1: phi = basis @ Wb  (C/D layout accumulators) ----
    v8f acc[4];
#pragma unroll
    for (int nt = 0; nt < 4; ++nt) {
        v8f c = {};
#pragma unroll
        for (int kh = 0; kh < 2; ++kh)
            c = WMMA(afrag[kh], load_bfrag(&wfrag[0][0], nt, kh, lane), c);
        acc[nt] = c;
    }

    // ---- inter = phi + bb + pp1[idx_i] + pp1[idx_j] ----
    int iidx[8], jidx[8];
#pragma unroll
    for (int v = 0; v < 8; ++v) {
        int em = base + v + 8 * hi; if (em >= Ecnt) em = Ecnt - 1;
        iidx[v] = idx_i[em]; jidx[v] = idx_j[em];
    }
#pragma unroll
    for (int nt = 0; nt < 4; ++nt)
#pragma unroll
        for (int v = 0; v < 8; ++v)
            acc[nt][v] += bbv[nt] + pp1[(size_t)iidx[v] * 64 + nt * 16 + lo]
                                  + pp1[(size_t)jidx[v] * 64 + nt * 16 + lo];

    // ---- transpose-stage inter, GEMM2: pi1 = tanh(inter @ Wpi + bpi) ----
#pragma unroll
    for (int nt = 0; nt < 4; ++nt)
#pragma unroll
        for (int v = 0; v < 8; ++v)
            Sm[(v + 8 * hi) * S_ROW + nt * 16 + lo] = (_Float16)acc[nt][v];
    __builtin_amdgcn_wave_barrier();
#pragma unroll
    for (int kh = 0; kh < 2; ++kh) afrag[kh] = load_afrag_from_S(Sm, lo, hi, kh);

#pragma unroll
    for (int nt = 0; nt < 4; ++nt) {
        v8f c = {};
#pragma unroll
        for (int kh = 0; kh < 2; ++kh)
            c = WMMA(afrag[kh], load_bfrag(&wfrag[1][0], nt, kh, lane), c);
        acc[nt] = c;
    }

    // ---- transpose-stage tanh(pi1), GEMM3: ii1 = tanh(pi1 @ Wii + bii) ----
    __builtin_amdgcn_wave_barrier();   // all fragment loads done before overwrite
#pragma unroll
    for (int nt = 0; nt < 4; ++nt)
#pragma unroll
        for (int v = 0; v < 8; ++v)
            Sm[(v + 8 * hi) * S_ROW + nt * 16 + lo] = (_Float16)TANH(acc[nt][v] + bpiv[nt]);
    __builtin_amdgcn_wave_barrier();
#pragma unroll
    for (int kh = 0; kh < 2; ++kh) afrag[kh] = load_afrag_from_S(Sm, lo, hi, kh);

#pragma unroll
    for (int nt = 0; nt < 4; ++nt) {
        v8f c = {};
#pragma unroll
        for (int kh = 0; kh < 2; ++kh)
            c = WMMA(afrag[kh], load_bfrag(&wfrag[2][0], nt, kh, lane), c);
        acc[nt] = c;
    }

    // ---- scatter_add: hardware fp32 atomics, lanes 0-15 hit 64B-contiguous run ----
#pragma unroll
    for (int nt = 0; nt < 4; ++nt)
#pragma unroll
        for (int v = 0; v < 8; ++v) {
            int em = base + v + 8 * hi;
            if (em < Ecnt) {
                float val = TANH(acc[nt][v] + biiv[nt]);
                unsafeAtomicAdd(&out[(size_t)iidx[v] * 64 + nt * 16 + lo], val);
            }
        }
}

// ---------------------------------------------------------------------------
extern "C" void kernel_launch(void* const* d_in, const int* in_sizes, int n_in,
                              void* d_out, int out_size, void* d_ws, size_t ws_size,
                              hipStream_t stream) {
    const int*   Z     = (const int*)  d_in[0];
    const float* R     = (const float*)d_in[1];
    const int*   idx_i = (const int*)  d_in[2];
    const int*   idx_j = (const int*)  d_in[3];
    const float* offs  = (const float*)d_in[4];
    const float* emb   = (const float*)d_in[5];
    const float* Wb    = (const float*)d_in[6];
    const float* bb    = (const float*)d_in[7];
    const float* Wpp1  = (const float*)d_in[8];
    const float* bpp1  = (const float*)d_in[9];
    const float* Wpp2  = (const float*)d_in[10];
    const float* bpp2  = (const float*)d_in[11];
    const float* Wpi   = (const float*)d_in[12];
    const float* bpi   = (const float*)d_in[13];
    const float* Wii   = (const float*)d_in[14];
    const float* bii   = (const float*)d_in[15];

    int N = in_sizes[0];
    int E = in_sizes[2];
    float* out = (float*)d_out;
    float* pp1 = (float*)d_ws;          // N*64 floats of scratch

    pinet_zero<<<(N * 64 + 255) / 256, 256, 0, stream>>>(out, N * 64);

    int ntiles = (N + 15) / 16;
    pinet_node<<<(ntiles + 7) / 8, 256, 0, stream>>>(Z, emb, Wpp1, bpp1, Wpp2, bpp2,
                                                     pp1, N, ntiles);

    int etiles = (E + 15) / 16;
    pinet_edge<<<(etiles + 7) / 8, 256, 0, stream>>>(R, idx_i, idx_j, offs,
                                                     Wb, bb, Wpi, bpi, Wii, bii,
                                                     pp1, out, E, etiles);
}
